// Simple_Cross2_13718125543492
// MI455X (gfx1250) — matrix operand order repaired and checked by reading, not compiled, not run on hardware
//
#include <hip/hip_runtime.h>

typedef __bf16 bf16;
typedef __attribute__((ext_vector_type(16))) __bf16 v16bf;
typedef __attribute__((ext_vector_type(8)))  __bf16 v8bf;
typedef __attribute__((ext_vector_type(8)))  float  v8f;

typedef __attribute__((address_space(3))) bf16 bf16_lds;

union FragBF { v16bf f; v8bf h[2]; };

#define BM 128
#define BN 128
#define BK 32
#define BKP 40   // padded LDS stride (80B = 20 banks -> conflict-free frag reads)
#define STAGES 3 // triple-buffered async DMA pipeline

// ---------------------------------------------------------------------------
// f32 -> bf16 conversion (weights, once per launch)
// ---------------------------------------------------------------------------
__global__ __launch_bounds__(256)
void cvt_f32_bf16(const float* __restrict__ src, bf16* __restrict__ dst, int n) {
    int i = blockIdx.x * 256 + threadIdx.x;
    if (i < n) dst[i] = (bf16)src[i];
}

// ---------------------------------------------------------------------------
// Cross layer: h0 = x * (x . cross_w) + cross_b + x   (bf16 output)
// one wave32 per row, 8 rows per block
// ---------------------------------------------------------------------------
__global__ __launch_bounds__(256)
void cross_kernel(const float* __restrict__ x, const float* __restrict__ cw,
                  const float* __restrict__ cb, bf16* __restrict__ h0, int F) {
    int row  = blockIdx.x * 8 + (threadIdx.x >> 5);
    int lane = threadIdx.x & 31;
    const float* xr = x + (size_t)row * F;
    float p = 0.f;
    for (int j = lane; j < F; j += 32) p += xr[j] * cw[j];
    #pragma unroll
    for (int m = 16; m >= 1; m >>= 1) p += __shfl_xor(p, m, 32);
    bf16* hr = h0 + (size_t)row * F;
    for (int j = lane; j < F; j += 32) {
        float v = xr[j];
        hr[j] = (bf16)(v * p + cb[j] + v);
    }
}

// ---------------------------------------------------------------------------
// out[M,N] = act( A[M,K](bf16,row-major) @ W[N,K]^T (bf16,row-major) + bias )
// 128x128 tile per block, 8 waves, wave = 64x32 patch = 8 x v_wmma 16x16x32.
// Tiles staged via CDNA5 GLOBAL_LOAD_ASYNC_TO_LDS_B128 (ASYNCcnt), 3-deep
// pipeline: DMA for tile s+2 issued while computing tile s; in-order async
// completion lets s_wait_asynccnt 0x4 retire exactly the oldest tile.
// ---------------------------------------------------------------------------
__global__ __launch_bounds__(256)
void gemm_bias_relu_bf16(const bf16* __restrict__ A,
                         const bf16* __restrict__ W,
                         const float* __restrict__ bias,
                         bf16* __restrict__ out,
                         int N, int K, int relu) {
    __shared__ bf16 As[STAGES][BM * BKP];
    __shared__ bf16 Bs[STAGES][BN * BKP];

    const int tid  = threadIdx.x;
    const int lane = tid & 31;
    const int wave = tid >> 5;
    const int wm   = wave >> 2;   // 0..1  (64-row half)
    const int wn   = wave & 3;    // 0..3  (32-col quarter)

    const int mBase = blockIdx.y * BM;
    const int nBase = blockIdx.x * BN;

    v8f acc[4][2];
    #pragma unroll
    for (int i = 0; i < 4; ++i)
        #pragma unroll
        for (int j = 0; j < 2; ++j)
            #pragma unroll
            for (int r = 0; r < 8; ++r) acc[i][j][r] = 0.f;

    const int steps = K / BK;

    // per-thread chunk mapping: 512 chunks of 8 bf16 (16B), 256 threads x 2
    int c0row[2], c0kp[2];
    #pragma unroll
    for (int i = 0; i < 2; ++i) {
        int c = tid + i * 256;
        c0row[i] = c >> 2;
        c0kp[i]  = (c & 3) * 8;
    }

    // async global -> LDS tile copy: 4 x b128 per thread per tile
    auto asyncTile = [&](int s, int buf) {
        const bf16* Ag = A + (size_t)mBase * K + (size_t)s * BK;
        const bf16* Wg = W + (size_t)nBase * K + (size_t)s * BK;
        #pragma unroll
        for (int i = 0; i < 2; ++i) {
            const bf16* ga = Ag + (size_t)c0row[i] * K + c0kp[i];
            const bf16* gb = Wg + (size_t)c0row[i] * K + c0kp[i];
            unsigned la = (unsigned)(unsigned long long)(bf16_lds*)(&As[buf][c0row[i] * BKP + c0kp[i]]);
            unsigned lb = (unsigned)(unsigned long long)(bf16_lds*)(&Bs[buf][c0row[i] * BKP + c0kp[i]]);
            unsigned long long gau = (unsigned long long)ga;
            unsigned long long gbu = (unsigned long long)gb;
            asm volatile("global_load_async_to_lds_b128 %0, %1, off"
                         :: "v"(la), "v"(gau) : "memory");
            asm volatile("global_load_async_to_lds_b128 %0, %1, off"
                         :: "v"(lb), "v"(gbu) : "memory");
        }
    };

    // prologue: fill stages 0 and 1, wait until stage 0 landed (<=4 in flight)
    asyncTile(0, 0);
    if (steps > 1) {
        asyncTile(1, 1);
        asm volatile("s_wait_asynccnt 0x4" ::: "memory");
    } else {
        asm volatile("s_wait_asynccnt 0x0" ::: "memory");
    }
    __syncthreads();

    int cur = 0;
    for (int s = 0; s < steps; ++s) {
        if (s + 2 < steps) {
            int nb = cur + 2; if (nb >= STAGES) nb -= STAGES;
            asyncTile(s + 2, nb);     // DMA two tiles ahead
        }

        // fragment loads (CDNA5 wave32 WMMA layouts)
        FragBF af[4], bf[2];
        const int aK0 = (lane < 16) ? 0 : 8;    // A: lanes 0-15 K{0..7,16..23}
        const int bK0 = (lane < 16) ? 0 : 8;    // B: lanes 0-15 K{0..7,16..23}
        #pragma unroll
        for (int mt = 0; mt < 4; ++mt) {
            int row = wm * 64 + mt * 16 + (lane & 15);
            af[mt].h[0] = *(const v8bf*)(&As[cur][row * BKP + aK0]);
            af[mt].h[1] = *(const v8bf*)(&As[cur][row * BKP + aK0 + 16]);
        }
        #pragma unroll
        for (int nt = 0; nt < 2; ++nt) {
            int row = wn * 32 + nt * 16 + (lane & 15);
            bf[nt].h[0] = *(const v8bf*)(&Bs[cur][row * BKP + bK0]);
            bf[nt].h[1] = *(const v8bf*)(&Bs[cur][row * BKP + bK0 + 16]);
        }

        #pragma unroll
        for (int mt = 0; mt < 4; ++mt)
            #pragma unroll
            for (int nt = 0; nt < 2; ++nt)
                acc[mt][nt] = __builtin_amdgcn_wmma_f32_16x16x32_bf16(
                    false, af[mt].f, false, bf[nt].f,
                    (short)0, acc[mt][nt], false, false);

        if (s + 1 < steps) {
            if (s + 2 < steps)
                asm volatile("s_wait_asynccnt 0x4" ::: "memory"); // tile s+1 landed
            else
                asm volatile("s_wait_asynccnt 0x0" ::: "memory");
            __syncthreads();
            ++cur; if (cur >= STAGES) cur = 0;
        }
    }

    // epilogue: bias + relu + bf16 store (C/D layout: vgpr r -> M=r | M=8+r)
    const int colBase = nBase + wn * 32 + (lane & 15);
    #pragma unroll
    for (int nt = 0; nt < 2; ++nt) {
        int col = colBase + nt * 16;
        float b = bias[col];
        #pragma unroll
        for (int mt = 0; mt < 4; ++mt) {
            int rowBase = mBase + wm * 64 + mt * 16 + ((lane < 16) ? 0 : 8);
            #pragma unroll
            for (int r = 0; r < 8; ++r) {
                float v = acc[mt][nt][r] + b;
                if (relu) v = v > 0.f ? v : 0.f;
                out[(size_t)(rowBase + r) * N + col] = (bf16)v;
            }
        }
    }
}

// ---------------------------------------------------------------------------
// Final layer: out[b] = h4[b,:] . Wo + bo     (one wave32 per row)
// ---------------------------------------------------------------------------
__global__ __launch_bounds__(256)
void final_kernel(const bf16* __restrict__ h4, const bf16* __restrict__ wo,
                  const float* __restrict__ bo, float* __restrict__ out, int K) {
    int row  = blockIdx.x * 8 + (threadIdx.x >> 5);
    int lane = threadIdx.x & 31;
    const bf16* hr = h4 + (size_t)row * K;
    float p = 0.f;
    for (int j = lane; j < K; j += 32) p += (float)hr[j] * (float)wo[j];
    #pragma unroll
    for (int m = 16; m >= 1; m >>= 1) p += __shfl_xor(p, m, 32);
    if (lane == 0) out[row] = p + bo[0];
}

// ---------------------------------------------------------------------------
extern "C" void kernel_launch(void* const* d_in, const int* in_sizes, int n_in,
                              void* d_out, int out_size, void* d_ws, size_t ws_size,
                              hipStream_t stream) {
    const int B = 16384, IN = 512, H1 = 2048, H2 = 2048, H3 = 1024, H4 = 512;

    const float* x   = (const float*)d_in[0];
    const float* cw  = (const float*)d_in[1];
    const float* cb  = (const float*)d_in[2];
    const float* W1  = (const float*)d_in[3];
    const float* b1  = (const float*)d_in[4];
    const float* W2  = (const float*)d_in[5];
    const float* b2  = (const float*)d_in[6];
    const float* W3  = (const float*)d_in[7];
    const float* b3  = (const float*)d_in[8];
    const float* W4  = (const float*)d_in[9];
    const float* b4  = (const float*)d_in[10];
    const float* Wo  = (const float*)d_in[11];
    const float* bo  = (const float*)d_in[12];
    float* out = (float*)d_out;

    // workspace carve-up (bf16 buffers), 256B aligned
    char* ws = (char*)d_ws;
    size_t off = 0;
    auto carve = [&](size_t elems) {
        bf16* p = (bf16*)(ws + off);
        off += ((elems * sizeof(bf16) + 255) & ~(size_t)255);
        return p;
    };
    bf16* h0  = carve((size_t)B * IN);
    bf16* hA  = carve((size_t)B * H1);   // h1, later h3
    bf16* hB  = carve((size_t)B * H2);   // h2, later h4
    bf16* W1b = carve((size_t)H1 * IN);
    bf16* W2b = carve((size_t)H2 * H1);
    bf16* W3b = carve((size_t)H3 * H2);
    bf16* W4b = carve((size_t)H4 * H3);
    bf16* Wob = carve((size_t)H4);

    // 1) convert weights to bf16
    auto cvt = [&](const float* s, bf16* d, int n) {
        cvt_f32_bf16<<<(n + 255) / 256, 256, 0, stream>>>(s, d, n);
    };
    cvt(W1, W1b, H1 * IN);
    cvt(W2, W2b, H2 * H1);
    cvt(W3, W3b, H3 * H2);
    cvt(W4, W4b, H4 * H3);
    cvt(Wo, Wob, H4);

    // 2) cross layer -> h0 (bf16)
    cross_kernel<<<B / 8, 256, 0, stream>>>(x, cw, cb, h0, IN);

    // 3) GEMM chain (bias + relu, bf16 activations)
    dim3 blk(256);
    gemm_bias_relu_bf16<<<dim3(H1 / BN, B / BM), blk, 0, stream>>>(h0, W1b, b1, hA, H1, IN, 1);
    gemm_bias_relu_bf16<<<dim3(H2 / BN, B / BM), blk, 0, stream>>>(hA, W2b, b2, hB, H2, H1, 1);
    gemm_bias_relu_bf16<<<dim3(H3 / BN, B / BM), blk, 0, stream>>>(hB, W3b, b3, hA, H3, H2, 1);
    gemm_bias_relu_bf16<<<dim3(H4 / BN, B / BM), blk, 0, stream>>>(hA, W4b, b4, hB, H4, H3, 1);

    // 4) output head
    final_kernel<<<B / 8, 256, 0, stream>>>(hB, Wob, bo, out, H4);
}
